// UniPhyBlock_3100966388307
// MI455X (gfx1250) — matrix operand order repaired and checked by reading, not compiled, not run on hardware
//
#include <hip/hip_runtime.h>

typedef __attribute__((ext_vector_type(16))) __bf16 v16bf;
typedef __attribute__((ext_vector_type(8)))  __bf16 v8bf;
typedef __attribute__((ext_vector_type(8)))  float  v8f;

// ---------------- problem constants ----------------
constexpr int Bc = 2, Tc = 8, Dc = 64, Hc = 64, Wc = 64;
constexpr int D2 = 128, Ec = 4, HIDc = 256;
constexpr int HW   = Hc * Wc;            // 4096
constexpr int NPIX = Bc * Tc * HW;       // 65536
constexpr int BT   = Bc * Tc;            // 16
constexpr int BTD  = BT * Dc;            // 1024

#define DEV static __device__ __forceinline__

union BF2 { __bf16 b[2]; unsigned int u; };

// ---------------- math helpers ----------------
DEV float softplusf(float x) { return fmaxf(x, 0.f) + log1pf(expf(-fabsf(x))); }
DEV float sigmoidf(float x)  { return 1.f / (1.f + expf(-x)); }
DEV float gelu_tanh(float x) {
    float x3 = x * x * x;
    return 0.5f * x * (1.f + tanhf(0.7978845608028654f * (x + 0.044715f * x3)));
}

// ---------------- WMMA helpers (wave32, 16x16x32 bf16, fp32 acc) ----------------
DEV v8f wmma_bf16(v16bf a, v16bf b, v8f c) {
    return __builtin_amdgcn_wmma_f32_16x16x32_bf16(false, a, false, b, (short)0, c, false, false);
}

// A fragment 16x32 bf16: lane (m = l&15, kh = l>>4) reads two contiguous 16B chunks:
// a[0..15] = row[kh*16 .. +15]   (matches swizzled B: k = 16*kh + s)
DEV v16bf frag_A_row(const __bf16* row) {
    int lane = threadIdx.x & 31;
    int kh = (lane >> 4) & 1;
    v8bf lo = *(const v8bf*)(row + kh * 16);
    v8bf hi = *(const v8bf*)(row + kh * 16 + 8);
    return __builtin_shufflevector(lo, hi, 0,1,2,3,4,5,6,7,8,9,10,11,12,13,14,15);
}
DEV v16bf frag_A(const __bf16* tile, int ld) {
    int lane = threadIdx.x & 31;
    return frag_A_row(tile + (size_t)(lane & 15) * ld);
}
// B fragment from pre-swizzled weights: each lane's 16 values stored contiguously
// fragbase layout: [lane 0..31][16 bf16]; lane l value m = W[kt*32 + (l>>4)*16 + m][nt*16 + (l&15)]
DEV v16bf frag_B_swz(const __bf16* fragbase) {
    int lane = threadIdx.x & 31;
    const __bf16* p = fragbase + lane * 16;
    v8bf lo = *(const v8bf*)p;
    v8bf hi = *(const v8bf*)(p + 8);
    return __builtin_shufflevector(lo, hi, 0,1,2,3,4,5,6,7,8,9,10,11,12,13,14,15);
}
// C/D fragment: lane 0-15 N=lane, VGPR j holds M=j (lanes<16) / M=j+8 (lanes>=16)
DEV void store_C_tile(float* dst, int ld, v8f c) {
    int lane = threadIdx.x & 31;
    int n  = lane & 15;
    int mh = (lane >> 4) & 1;
#pragma unroll
    for (int j = 0; j < 8; ++j) dst[(size_t)(mh * 8 + j) * ld + n] = c[j];
}
DEV void store_C_gelu_bf16(__bf16* dst, int ld, v8f c, const float* bias16) {
    int lane = threadIdx.x & 31;
    int n  = lane & 15;
    int mh = (lane >> 4) & 1;
    float bn = bias16[n];
#pragma unroll
    for (int j = 0; j < 8; ++j) {
        float v = gelu_tanh(c[j] + bn);
        dst[(size_t)(mh * 8 + j) * ld + n] = (__bf16)v;
    }
}

// ---------------- block reduction ----------------
DEV float block_sum(float v, float* sbuf) {
    int tid = threadIdx.x;
    sbuf[tid] = v;
    __syncthreads();
    for (int s = 128; s > 0; s >>= 1) {
        if (tid < s) sbuf[tid] += sbuf[tid + s];
        __syncthreads();
    }
    float r = sbuf[0];
    __syncthreads();
    return r;
}

// ================= K0: weight prep -> bf16 fragment-swizzled layouts =================
// swizzled layout per matrix (KxN): frag (nt,kt) at offset ((nt*(K/32)+kt)*32+lane)*16
__global__ __launch_bounds__(256) void prep_weights_kernel(
    const float* conv_w, const float* enc_re, const float* enc_im,
    const float* dec_re, const float* dec_im, const float* w1, const float* w2,
    __bf16* cw_s, __bf16* enc_s, __bf16* dec_s, __bf16* w1_s, __bf16* w2_s)
{
    int i = blockIdx.x * 256 + threadIdx.x;           // lane-row id
    const int LR_CONV = 9 * 1024;                      // K=128,N=128 per tap
    const int LR_CAT  = 1024;
    const int LR_W1   = Ec * 2048;                     // K=128,N=256 per expert
    const int LR_W2   = Ec * 2048;                     // K=256,N=128 per expert
    __bf16 vals[16];
    __bf16* dst;

    if (i < LR_CONV) {
        int lr = i & 1023;
        int lane = lr & 31, tl = lr >> 5;              // KT=4
        int kt = tl & 3, nt = tl >> 2;
        int n = nt * 16 + (lane & 15);
        int kb = kt * 32 + (lane >> 4) * 16;
        const float* src = conv_w + (i >> 10) * 16384;
#pragma unroll
        for (int m = 0; m < 16; ++m) vals[m] = (__bf16)src[(kb + m) * 128 + n];
        dst = cw_s + (size_t)i * 16;
    } else if ((i -= LR_CONV) < 2 * LR_CAT) {
        bool isdec = i >= LR_CAT;
        int lr = i & 1023;
        int lane = lr & 31, tl = lr >> 5;              // KT=4
        int kt = tl & 3, nt = tl >> 2;
        int n = nt * 16 + (lane & 15);
        int kb = kt * 32 + (lane >> 4) * 16;
        const float* wre = isdec ? dec_re : enc_re;
        const float* wim = isdec ? dec_im : enc_im;
#pragma unroll
        for (int m = 0; m < 16; ++m) {
            int k = kb + m;
            float v;
            if (k < 64) v = (n < 64) ? wre[k * 64 + n] : wim[k * 64 + (n - 64)];
            else        v = (n < 64) ? -wim[(k - 64) * 64 + n] : wre[(k - 64) * 64 + (n - 64)];
            vals[m] = (__bf16)v;
        }
        dst = (isdec ? dec_s : enc_s) + (size_t)lr * 16;
    } else if ((i -= 2 * LR_CAT) < LR_W1) {
        int lr = i & 2047;
        int lane = lr & 31, tl = lr >> 5;              // KT=4, NT=16
        int kt = tl & 3, nt = tl >> 2;
        int n = nt * 16 + (lane & 15);
        int kb = kt * 32 + (lane >> 4) * 16;
        const float* src = w1 + (i >> 11) * (D2 * HIDc);
#pragma unroll
        for (int m = 0; m < 16; ++m) vals[m] = (__bf16)src[(kb + m) * HIDc + n];
        dst = w1_s + (size_t)i * 16;
    } else if ((i -= LR_W1) < LR_W2) {
        int lr = i & 2047;
        int lane = lr & 31, tl = lr >> 5;              // KT=8, NT=8
        int kt = tl & 7, nt = tl >> 3;
        int n = nt * 16 + (lane & 15);
        int kb = kt * 32 + (lane >> 4) * 16;
        const float* src = w2 + (i >> 11) * (HIDc * D2);
#pragma unroll
        for (int m = 0; m < 16; ++m) vals[m] = (__bf16)src[(kb + m) * D2 + n];
        dst = w2_s + (size_t)i * 16;
    } else return;

#pragma unroll
    for (int m = 0; m < 16; ++m) dst[m] = vals[m];
}

// ================= K1: per (b,t,d) stats over HxW =================
__global__ __launch_bounds__(256) void stats_kernel(
    const float* xre, const float* xim, float* te, float* mre, float* mim)
{
    __shared__ float sbuf[256];
    int btd = blockIdx.x;
    const float* pr = xre + (size_t)btd * HW;
    const float* pi = xim + (size_t)btd * HW;
    float se = 0.f, sr = 0.f, si = 0.f;
    for (int i = threadIdx.x; i < HW; i += 256) {
        float a = pr[i], b = pi[i];
        se += a * a + b * b; sr += a; si += b;
    }
    se = block_sum(se, sbuf);
    sr = block_sum(sr, sbuf);
    si = block_sum(si, sbuf);
    if (threadIdx.x == 0) {
        te[btd]  = se * (1.f / HW);
        mre[btd] = sr * (1.f / HW);
        mim[btd] = si * (1.f / HW);
    }
}

// ================= K2: prelude (momentum GEMM, operators) =================
__global__ __launch_bounds__(256) void prelude_kernel(
    const float* dt, const float* mom_w, const float* mom_b, const float* adv_w,
    const float* lam_re, const float* lam_im,
    const float* flux_a_re, const float* flux_a_im,
    const float* mre, const float* mim,
    float* drift_re, float* drift_im,
    float* opd_re, float* opd_im, float* opf_re, float* opf_im,
    float* af_re, float* af_im)
{
    for (int idx = threadIdx.x; idx < BTD; idx += 256) {
        int bt = idx >> 6, d = idx & 63, t = bt & 7;
        float sre = mom_b[d], sim = mom_b[64 + d];
        for (int k = 0; k < 64; ++k) {
            float a = mre[bt * 64 + k], b = mim[bt * 64 + k];
            sre += a * mom_w[k * D2 + d]      + b * mom_w[(64 + k) * D2 + d];
            sim += a * mom_w[k * D2 + 64 + d] + b * mom_w[(64 + k) * D2 + 64 + d];
        }
        float sc = adv_w[0] * dt[t];
        drift_re[idx] = sre * sc;
        drift_im[idx] = sim * sc;
    }
    for (int idx = threadIdx.x; idx < Tc * Dc; idx += 256) {
        int t = idx >> 6, d = idx & 63;
        float lr = -softplusf(lam_re[d]), li = lam_im[d];
        float zr = lr * dt[t], zi = li * dt[t];
        float e = expf(zr);
        float dr_ = e * cosf(zi), di_ = e * sinf(zi);
        opd_re[idx] = dr_; opd_im[idx] = di_;
        float nr = dr_ - 1.f, ni = di_;
        float den = lr * lr + li * li;
        opf_re[idx] = (nr * lr + ni * li) / den;
        opf_im[idx] = (ni * lr - nr * li) / den;
    }
    for (int d = threadIdx.x; d < Dc; d += 256) {
        float ar = -softplusf(flux_a_re[d]), ai = flux_a_im[d];
        float e = expf(ar);
        af_re[d] = e * cosf(ai);
        af_im[d] = e * sinf(ai);
    }
}

// ================= K3: fused LayerNorm + 3x3 conv (implicit GEMM) + residual =================
__global__ __launch_bounds__(256) void ln_conv_kernel(
    const float* xre, const float* xim, const float* ln_g, const float* ln_b,
    const __bf16* cw_s, const float* conv_b,
    const float* drift_re, const float* drift_im,
    float* xcl_re, float* xcl_im, __bf16* xcat)
{
    __shared__ __align__(16) __bf16 shin[10 * 10 * D2];
    __shared__ __align__(16) float  shy[64 * D2];

    int blk  = blockIdx.x;
    int bt   = blk >> 6;
    int tile = blk & 63;
    int th0 = (tile >> 3) * 8, tw0 = (tile & 7) * 8;
    const float* xre_bt = xre + (size_t)bt * Dc * HW;
    const float* xim_bt = xim + (size_t)bt * Dc * HW;

    // stage layernormed halo (zeros outside = SAME zero padding), packed pair stores
    for (int hp = threadIdx.x; hp < 100; hp += 256) {
        int hy = hp / 10, hx = hp % 10;
        int h = th0 + hy - 1, w = tw0 + hx - 1;
        unsigned int* dstu = (unsigned int*)&shin[hp * D2];
        if (h < 0 || h >= Hc || w < 0 || w >= Wc) {
            for (int c = 0; c < 64; ++c) dstu[c] = 0u;
        } else {
            int pix = h * Wc + w;
            float s1 = 0.f, s2 = 0.f;
            for (int d = 0; d < Dc; ++d) { float a = xre_bt[d * HW + pix]; s1 += a; s2 += a * a; }
            for (int d = 0; d < Dc; ++d) { float a = xim_bt[d * HW + pix]; s1 += a; s2 += a * a; }
            float m   = s1 * (1.f / D2);
            float var = s2 * (1.f / D2) - m * m;
            float inv = rsqrtf(var + 1e-5f);
            for (int d = 0; d < 64; d += 2) {
                BF2 p2;
                p2.b[0] = (__bf16)((xre_bt[d * HW + pix] - m) * inv * ln_g[d] + ln_b[d]);
                p2.b[1] = (__bf16)((xre_bt[(d + 1) * HW + pix] - m) * inv * ln_g[d + 1] + ln_b[d + 1]);
                dstu[d >> 1] = p2.u;
            }
            for (int d = 0; d < 64; d += 2) {
                BF2 p2;
                p2.b[0] = (__bf16)((xim_bt[d * HW + pix] - m) * inv * ln_g[64 + d] + ln_b[64 + d]);
                p2.b[1] = (__bf16)((xim_bt[(d + 1) * HW + pix] - m) * inv * ln_g[64 + d + 1] + ln_b[64 + d + 1]);
                dstu[(64 + d) >> 1] = p2.u;
            }
        }
    }
    __syncthreads();

    // conv = 9 shifted GEMMs; M=64 px (4 tiles), N=128 (8 tiles), K=128 per tap
    int wave = threadIdx.x >> 5;
    int lane = threadIdx.x & 31;
    int mt = wave & 3, nt0 = (wave >> 2) * 4;
    int p = mt * 16 + (lane & 15);
    int py = p >> 3, px = p & 7;

    v8f acc[4] = {};
    for (int tap = 0; tap < 9; ++tap) {
        int kh2 = tap / 3, kw2 = tap % 3;
        const __bf16* arow = &shin[((py + kh2) * 10 + (px + kw2)) * D2];
        const __bf16* btap = cw_s + (size_t)tap * 16384;
        __builtin_prefetch(btap + (size_t)(nt0 * 4) * 512, 0, 1);
#pragma unroll
        for (int kk = 0; kk < 4; ++kk) {
            v16bf a = frag_A_row(arow + kk * 32);
#pragma unroll
            for (int i = 0; i < 4; ++i) {
                v16bf b = frag_B_swz(btap + (size_t)((nt0 + i) * 4 + kk) * 512);
                acc[i] = wmma_bf16(a, b, acc[i]);
            }
        }
    }
#pragma unroll
    for (int i = 0; i < 4; ++i)
        store_C_tile(&shy[(size_t)(mt * 16) * D2 + (nt0 + i) * 16], D2, acc[i]);
    __syncthreads();

    // epilogue: xcl = x + conv + drift ; paired stores
    int n_base = bt * HW;
    for (int idx = threadIdx.x; idx < 64 * 64; idx += 256) {
        int pp = idx >> 6;
        int cp = (idx & 63) * 2;                 // channels cp, cp+1 (same half)
        int h = th0 + (pp >> 3), w = tw0 + (pp & 7);
        int pix = h * Wc + w;
        size_t n = (size_t)(n_base + pix);
        bool isre = cp < 64;
        int d = cp & 63;
        const float* xb  = isre ? xre_bt : xim_bt;
        const float* drp = isre ? drift_re : drift_im;
        float v0 = xb[d * HW + pix]       + shy[pp * D2 + cp]     + conv_b[cp]     + drp[bt * 64 + d];
        float v1 = xb[(d + 1) * HW + pix] + shy[pp * D2 + cp + 1] + conv_b[cp + 1] + drp[bt * 64 + d + 1];
        float* xclp = isre ? xcl_re : xcl_im;
        *(float2*)&xclp[n * Dc + d] = make_float2(v0, v1);
        BF2 p2; p2.b[0] = (__bf16)v0; p2.b[1] = (__bf16)v1;
        ((unsigned int*)xcat)[(n * D2 + cp) >> 1] = p2.u;
    }
}

// ================= K4/K8: (M x 128) @ (128 x 128) bf16 GEMM, swizzled B from L2 =================
__global__ __launch_bounds__(256) void gemm_nk128_kernel(
    const __bf16* A, const __bf16* Bs, float* C)
{
    __shared__ __align__(16) __bf16 shA[64 * D2];
    int m0 = blockIdx.x * 64;

    const uint4* srcA = (const uint4*)(A + (size_t)m0 * D2);
    for (int i = threadIdx.x; i < 1024; i += 256) ((uint4*)shA)[i] = srcA[i];
    __syncthreads();

    int wave = threadIdx.x >> 5;
    int mt = wave & 3, nt0 = (wave >> 2) * 4;
    __builtin_prefetch(Bs + (size_t)(nt0 * 4) * 512, 0, 1);
    v8f acc[4] = {};
#pragma unroll
    for (int kk = 0; kk < 4; ++kk) {
        v16bf a = frag_A(&shA[(size_t)(mt * 16) * D2 + kk * 32], D2);
#pragma unroll
        for (int i = 0; i < 4; ++i) {
            v16bf b = frag_B_swz(Bs + (size_t)((nt0 + i) * 4 + kk) * 512);
            acc[i] = wmma_bf16(a, b, acc[i]);
        }
    }
#pragma unroll
    for (int i = 0; i < 4; ++i)
        store_C_tile(C + (size_t)(m0 + mt * 16) * D2 + (nt0 + i) * 16, D2, acc[i]);
}

// ================= K5: x_mean reduction over pixels =================
__global__ __launch_bounds__(256) void reduce_xmean_kernel(
    const float* xeig, float* xmre, float* xmim)
{
    __shared__ float sbuf[256];
    int blk = blockIdx.x;
    int bt = blk >> 7, c = blk & 127;
    const float* base = xeig + (size_t)bt * HW * D2 + c;
    float s = 0.f;
    for (int p = threadIdx.x; p < HW; p += 256) s += base[(size_t)p * D2];
    s = block_sum(s, sbuf);
    if (threadIdx.x == 0) {
        s *= (1.f / HW);
        if (c < 64) xmre[bt * 64 + c] = s;
        else        xmim[bt * 64 + (c - 64)] = s;
    }
}

// ================= K6: flux scan / gate / source (tiny, one block) =================
__global__ __launch_bounds__(256) void flux_kernel(
    const float* xmre, const float* xmim,
    const float* fin_re, const float* fin_im,
    const float* fproj_re, const float* fproj_im,
    const float* gate_w, const float* gate_b,
    const float* af_re, const float* af_im,
    float* gout, float* src_re, float* src_im)
{
    __shared__ float xfr[BT * Dc], xfi[BT * Dc], flr[BT * Dc], fli[BT * Dc];
    for (int idx = threadIdx.x; idx < BTD; idx += 256) {
        int bt = idx >> 6, j = idx & 63;
        float sr = 0.f, si = 0.f;
        for (int d = 0; d < Dc; ++d) {
            float ar = xmre[bt * 64 + d], ai = xmim[bt * 64 + d];
            float br = fin_re[d * 64 + j], bi = fin_im[d * 64 + j];
            sr += ar * br - ai * bi;
            si += ar * bi + ai * br;
        }
        xfr[idx] = sr; xfi[idx] = si;
    }
    __syncthreads();
    for (int idx = threadIdx.x; idx < Bc * Dc; idx += 256) {
        int b = idx >> 6, d = idx & 63;
        float ar = af_re[d], ai = af_im[d];
        float pr = 0.f, pi = 0.f;
        for (int t = 0; t < Tc; ++t) {
            int bt = b * Tc + t;
            float nr = ar * pr - ai * pi + xfr[bt * 64 + d];
            float ni = ar * pi + ai * pr + xfi[bt * 64 + d];
            flr[bt * 64 + d] = nr; fli[bt * 64 + d] = ni;
            pr = nr; pi = ni;
        }
    }
    __syncthreads();
    for (int idx = threadIdx.x; idx < BTD; idx += 256) {
        int bt = idx >> 6, j = idx & 63;
        float sr = 0.f, si = 0.f, gl = gate_b[j];
        for (int d = 0; d < Dc; ++d) {
            float ar = flr[bt * 64 + d], ai = fli[bt * 64 + d];
            float br = fproj_re[d * 64 + j], bi = fproj_im[d * 64 + j];
            sr += ar * br - ai * bi;
            si += ar * bi + ai * br;
            gl += ar * gate_w[d * 64 + j] + ai * gate_w[(64 + d) * 64 + j];
        }
        src_re[idx] = sr; src_im[idx] = si;
        gout[idx] = sigmoidf(gl);
    }
}

// ================= K7: forcing + time scan + noise (channel-pair vectorized) =================
__global__ __launch_bounds__(256) void ut_scan_kernel(
    const float* xeig, const float* g, const float* src_re, const float* src_im,
    const float* opf_re, const float* opf_im, const float* opd_re, const float* opd_im,
    const float* dt, const float* noise_re, const float* noise_im,
    __bf16* ucat)
{
    int gid = blockIdx.x * 256 + threadIdx.x;      // (b, pix, dpair)
    if (gid >= Bc * HW * 32) return;
    int d = (gid & 31) * 2;
    int rest = gid >> 5;
    int p = rest & (HW - 1);
    int b = rest >> 12;

    float cr0 = 0.f, ci0 = 0.f, cr1 = 0.f, ci1 = 0.f;
    for (int t = 0; t < Tc; ++t) {
        int bt = b * Tc + t;
        size_t n = (size_t)bt * HW + p;
        float2 xer = *(const float2*)&xeig[n * D2 + d];
        float2 xei = *(const float2*)&xeig[n * D2 + 64 + d];
        float2 nre = *(const float2*)&noise_re[n * Dc + d];
        float2 nim = *(const float2*)&noise_im[n * Dc + d];
        float ns = 0.01f * sqrtf(dt[t]);
        BF2 pre, pim;
#pragma unroll
        for (int q = 0; q < 2; ++q) {
            int dd = d + q;
            float gg = g[bt * 64 + dd];
            float fr = (q ? xer.y : xer.x) * gg + src_re[bt * 64 + dd] * (1.f - gg);
            float fi = (q ? xei.y : xei.x) * gg + src_im[bt * 64 + dd] * (1.f - gg);
            float ofr = opf_re[t * 64 + dd], ofi = opf_im[t * 64 + dd];
            float ur = fr * ofr - fi * ofi;
            float ui = fr * ofi + fi * ofr;
            float adr = opd_re[t * 64 + dd], adi = opd_im[t * 64 + dd];
            float& cr = q ? cr1 : cr0;
            float& ci = q ? ci1 : ci0;
            float nr = adr * cr - adi * ci + ur;
            float ni = adr * ci + adi * cr + ui;
            cr = nr; ci = ni;
            pre.b[q] = (__bf16)(cr + ns * (q ? nre.y : nre.x));
            pim.b[q] = (__bf16)(ci + ns * (q ? nim.y : nim.x));
        }
        ((unsigned int*)ucat)[(n * D2 + d) >> 1]      = pre.u;
        ((unsigned int*)ucat)[(n * D2 + 64 + d) >> 1] = pim.u;
    }
}

// ================= K9: energy reduction -> scale factor =================
__global__ __launch_bounds__(256) void reduce_energy_kernel(
    const float* xout, const float* te, const float* e_scale, float* mult)
{
    __shared__ float sbuf[256];
    int blk = blockIdx.x;
    int bt = blk >> 6, d = blk & 63;
    const float* base = xout + (size_t)bt * HW * D2;
    float s = 0.f;
    for (int p = threadIdx.x; p < HW; p += 256) {
        float a = base[(size_t)p * D2 + d];
        float b = base[(size_t)p * D2 + 64 + d];
        s += a * a + b * b;
    }
    s = block_sum(s, sbuf);
    if (threadIdx.x == 0) {
        float cur = s * (1.f / HW);
        float ratio = te[blk] / (cur + 1e-8f);
        ratio = fminf(fmaxf(ratio, 0.5f), 2.f);
        float sf = sqrtf(ratio);
        mult[blk] = 1.f + e_scale[0] * (sf - 1.f);
    }
}

// ================= K10: fused scale + LN + router + dense MoE + residual + transpose =================
__global__ __launch_bounds__(256) void final_moe_kernel(
    const float* xout, const float* mult, const float* ln_g, const float* ln_b,
    const float* router_w, const float* router_b,
    const __bf16* w1s, const float* b1, const __bf16* w2s, const float* b2,
    const float* xcl_re, const float* xcl_im,
    float* out)
{
    __shared__ __align__(16) float  tokf[32 * D2];
    __shared__ __align__(16) __bf16 tokb[32 * D2];
    __shared__ __align__(16) __bf16 hmid[32 * HIDc];
    __shared__ float pexp[32 * Ec];

    int n0 = blockIdx.x * 32;

    for (int tl = threadIdx.x; tl < 32; tl += 256) {
        int n = n0 + tl;
        int bt = n >> 12;
        const float* row = xout + (size_t)n * D2;
        float s1 = 0.f, s2 = 0.f;
        for (int c = 0; c < D2; ++c) {
            float v = row[c] * mult[bt * 64 + (c & 63)];
            tokf[tl * D2 + c] = v;
            s1 += v; s2 += v * v;
        }
        float m = s1 * (1.f / D2);
        float var = s2 * (1.f / D2) - m * m;
        float inv = rsqrtf(var + 1e-5f);
        for (int c = 0; c < D2; ++c) {
            float v = (tokf[tl * D2 + c] - m) * inv * ln_g[c] + ln_b[c];
            tokf[tl * D2 + c] = v;
            tokb[tl * D2 + c] = (__bf16)v;
        }
    }
    __syncthreads();

    for (int idx = threadIdx.x; idx < 32 * Ec; idx += 256) {
        int tl = idx >> 2, e = idx & 3;
        float s = router_b[e];
        for (int c = 0; c < D2; ++c) s += tokf[tl * D2 + c] * router_w[c * Ec + e];
        pexp[idx] = s;
    }
    __syncthreads();
    for (int tl = threadIdx.x; tl < 32; tl += 256) {
        float* pr = &pexp[tl * Ec];
        float mx = fmaxf(fmaxf(pr[0], pr[1]), fmaxf(pr[2], pr[3]));
        float s = 0.f;
        for (int e = 0; e < Ec; ++e) { pr[e] = expf(pr[e] - mx); s += pr[e]; }
        float is = 1.f / s;
        for (int e = 0; e < Ec; ++e) pr[e] *= is;
    }

    int wave = threadIdx.x >> 5;
    int lane = threadIdx.x & 31;
    int nn = lane & 15, mh = (lane >> 4) & 1;

    int mt2 = wave & 1;
    int ntA = wave >> 1;            // second tile at ntA+4
    v8f accA = {}, accB = {};

    for (int e = 0; e < Ec; ++e) {
        __syncthreads();
        // GEMM1: (32x128)@(128x256) -> gelu -> hmid bf16 (swizzled w1, KT=4)
        int mt = wave & 1;
        int ntb = wave >> 1;
        const __bf16* w1e = w1s + (size_t)e * (D2 * HIDc);
        const float*  b1e = b1 + e * HIDc;
        __builtin_prefetch(w1e, 0, 1);
        v8f h[4] = {};
#pragma unroll
        for (int kk = 0; kk < 4; ++kk) {
            v16bf a = frag_A(&tokb[(size_t)(mt * 16) * D2 + kk * 32], D2);
#pragma unroll
            for (int i = 0; i < 4; ++i) {
                int nt = ntb + i * 4;
                v16bf b = frag_B_swz(w1e + (size_t)(nt * 4 + kk) * 512);
                h[i] = wmma_bf16(a, b, h[i]);
            }
        }
#pragma unroll
        for (int i = 0; i < 4; ++i) {
            int nt = ntb + i * 4;
            store_C_gelu_bf16(&hmid[(size_t)(mt * 16) * HIDc + nt * 16], HIDc, h[i], b1e + nt * 16);
        }
        __syncthreads();
        // GEMM2: (32x256)@(256x128) (swizzled w2, KT=8), p-weighted register accum
        const __bf16* w2e = w2s + (size_t)e * (HIDc * D2);
        const float*  b2e = b2 + e * D2;
        __builtin_prefetch(w2e, 0, 1);
        v8f c0 = {}, c1 = {};
#pragma unroll
        for (int kk = 0; kk < 8; ++kk) {
            v16bf a = frag_A(&hmid[(size_t)(mt2 * 16) * HIDc + kk * 32], HIDc);
            v16bf bf0 = frag_B_swz(w2e + (size_t)(ntA * 8 + kk) * 512);
            c0 = wmma_bf16(a, bf0, c0);
            v16bf bf1 = frag_B_swz(w2e + (size_t)((ntA + 4) * 8 + kk) * 512);
            c1 = wmma_bf16(a, bf1, c1);
        }
        float bA = b2e[ntA * 16 + nn];
        float bB = b2e[(ntA + 4) * 16 + nn];
#pragma unroll
        for (int j = 0; j < 8; ++j) {
            int ml = mt2 * 16 + mh * 8 + j;
            float pe = pexp[ml * Ec + e];
            accA[j] += pe * (c0[j] + bA);
            accB[j] += pe * (c1[j] + bB);
        }
    }

    // epilogue: out_tok = tok + moe ; z = xcl + complex(out) ; write (2,B,T,D,H,W)
#pragma unroll
    for (int part = 0; part < 2; ++part) {
        int nt = ntA + part * 4;
        int c = nt * 16 + nn;
        int d = c & 63;
        bool isre = (c < 64);
#pragma unroll
        for (int j = 0; j < 8; ++j) {
            int ml = mt2 * 16 + mh * 8 + j;
            int n = n0 + ml;
            float v = (part ? accB[j] : accA[j]);
            float outv = tokf[ml * D2 + c] + v;
            int wq = n & 63, hq = (n >> 6) & 63, tq = (n >> 12) & 7, bq = (n >> 15) & 1;
            float base = (isre ? xcl_re : xcl_im)[(size_t)n * Dc + d];
            int s = isre ? 0 : 1;
            size_t o = (((size_t)((s * Bc + bq) * Tc + tq) * Dc + d) * HW) + hq * Wc + wq;
            out[o] = base + outv;
        }
    }
}

// ================= host launcher =================
extern "C" void kernel_launch(void* const* d_in, const int* in_sizes, int n_in,
                              void* d_out, int out_size, void* d_ws, size_t ws_size,
                              hipStream_t stream)
{
    (void)in_sizes; (void)n_in; (void)out_size; (void)ws_size;

    const float* x_re      = (const float*)d_in[0];
    const float* x_im      = (const float*)d_in[1];
    const float* dt        = (const float*)d_in[2];
    const float* ln_s_g    = (const float*)d_in[3];
    const float* ln_s_b    = (const float*)d_in[4];
    const float* conv_w    = (const float*)d_in[5];
    const float* conv_b    = (const float*)d_in[6];
    const float* mom_w     = (const float*)d_in[7];
    const float* mom_b     = (const float*)d_in[8];
    const float* adv_w     = (const float*)d_in[9];
    const float* enc_re    = (const float*)d_in[10];
    const float* enc_im    = (const float*)d_in[11];
    const float* dec_re    = (const float*)d_in[12];
    const float* dec_im    = (const float*)d_in[13];
    const float* flux_a_re = (const float*)d_in[14];
    const float* flux_a_im = (const float*)d_in[15];
    const float* fin_re    = (const float*)d_in[16];
    const float* fin_im    = (const float*)d_in[17];
    const float* fproj_re  = (const float*)d_in[18];
    const float* fproj_im  = (const float*)d_in[19];
    const float* gate_w    = (const float*)d_in[20];
    const float* gate_b    = (const float*)d_in[21];
    const float* lam_re    = (const float*)d_in[22];
    const float* lam_im    = (const float*)d_in[23];
    const float* noise_re  = (const float*)d_in[24];
    const float* noise_im  = (const float*)d_in[25];
    const float* e_scale   = (const float*)d_in[26];
    const float* ln_t_g    = (const float*)d_in[27];
    const float* ln_t_b    = (const float*)d_in[28];
    const float* router_w  = (const float*)d_in[29];
    const float* router_b  = (const float*)d_in[30];
    const float* w1        = (const float*)d_in[31];
    const float* b1        = (const float*)d_in[32];
    const float* w2        = (const float*)d_in[33];
    const float* b2        = (const float*)d_in[34];

    char* wsb = (char*)d_ws;
    size_t off = 0;
    auto take = [&](size_t bytes) { char* p = wsb + off; off += (bytes + 255) & ~(size_t)255; return p; };

    __bf16* cw_s     = (__bf16*)take((size_t)9 * D2 * D2 * 2);
    __bf16* enc_s    = (__bf16*)take((size_t)D2 * D2 * 2);
    __bf16* dec_s    = (__bf16*)take((size_t)D2 * D2 * 2);
    __bf16* w1_s     = (__bf16*)take((size_t)Ec * D2 * HIDc * 2);
    __bf16* w2_s     = (__bf16*)take((size_t)Ec * HIDc * D2 * 2);
    float* te        = (float*)take(BTD * 4);
    float* mre       = (float*)take(BTD * 4);
    float* mim       = (float*)take(BTD * 4);
    float* drift_re  = (float*)take(BTD * 4);
    float* drift_im  = (float*)take(BTD * 4);
    float* opd_re    = (float*)take(Tc * Dc * 4);
    float* opd_im    = (float*)take(Tc * Dc * 4);
    float* opf_re    = (float*)take(Tc * Dc * 4);
    float* opf_im    = (float*)take(Tc * Dc * 4);
    float* af_re     = (float*)take(Dc * 4);
    float* af_im     = (float*)take(Dc * 4);
    float* xmre      = (float*)take(BTD * 4);
    float* xmim      = (float*)take(BTD * 4);
    float* gbuf      = (float*)take(BTD * 4);
    float* src_re    = (float*)take(BTD * 4);
    float* src_im    = (float*)take(BTD * 4);
    float* multb     = (float*)take(BTD * 4);
    float* xcl_re    = (float*)take((size_t)NPIX * Dc * 4);
    float* xcl_im    = (float*)take((size_t)NPIX * Dc * 4);
    __bf16* xcat     = (__bf16*)take((size_t)NPIX * D2 * 2);
    float* xeig      = (float*)take((size_t)NPIX * D2 * 4);
    __bf16* ucat     = (__bf16*)take((size_t)NPIX * D2 * 2);
    float* xout      = (float*)take((size_t)NPIX * D2 * 4);

    float* out = (float*)d_out;

    {
        int total_lr = 9 * 1024 + 2 * 1024 + Ec * 2048 + Ec * 2048;   // 27648
        prep_weights_kernel<<<(total_lr + 255) / 256, 256, 0, stream>>>(
            conv_w, enc_re, enc_im, dec_re, dec_im, w1, w2,
            cw_s, enc_s, dec_s, w1_s, w2_s);
    }
    stats_kernel<<<BTD, 256, 0, stream>>>(x_re, x_im, te, mre, mim);
    prelude_kernel<<<1, 256, 0, stream>>>(
        dt, mom_w, mom_b, adv_w, lam_re, lam_im, flux_a_re, flux_a_im,
        mre, mim, drift_re, drift_im, opd_re, opd_im, opf_re, opf_im, af_re, af_im);
    ln_conv_kernel<<<BT * 64, 256, 0, stream>>>(
        x_re, x_im, ln_s_g, ln_s_b, cw_s, conv_b, drift_re, drift_im,
        xcl_re, xcl_im, xcat);
    gemm_nk128_kernel<<<NPIX / 64, 256, 0, stream>>>(xcat, enc_s, xeig);
    reduce_xmean_kernel<<<BT * D2, 256, 0, stream>>>(xeig, xmre, xmim);
    flux_kernel<<<1, 256, 0, stream>>>(
        xmre, xmim, fin_re, fin_im, fproj_re, fproj_im, gate_w, gate_b,
        af_re, af_im, gbuf, src_re, src_im);
    ut_scan_kernel<<<(Bc * HW * 32) / 256, 256, 0, stream>>>(
        xeig, gbuf, src_re, src_im, opf_re, opf_im, opd_re, opd_im,
        dt, noise_re, noise_im, ucat);
    gemm_nk128_kernel<<<NPIX / 64, 256, 0, stream>>>(ucat, dec_s, xout);
    reduce_energy_kernel<<<BTD, 256, 0, stream>>>(xout, te, e_scale, multb);
    final_moe_kernel<<<NPIX / 32, 256, 0, stream>>>(
        xout, multb, ln_t_g, ln_t_b, router_w, router_b,
        w1_s, b1, w2_s, b2, xcl_re, xcl_im, out);
}
